// BlockAttention_64141041598498
// MI455X (gfx1250) — compile-verified, matrix operand order checked by reading, not run on hardware
//
#include <hip/hip_runtime.h>
#include <hip/hip_bf16.h>

typedef __attribute__((ext_vector_type(16))) __bf16 v16bf;
typedef __attribute__((ext_vector_type(8)))  float  v8f;

__device__ __forceinline__ __bf16 f2bf(float f) {
  unsigned u = __builtin_bit_cast(unsigned, f);
  u += 0x7FFFu + ((u >> 16) & 1u);              // round-to-nearest-even
  unsigned short h = (unsigned short)(u >> 16);
  return __builtin_bit_cast(__bf16, h);
}

__device__ __forceinline__ float geluf(float x) {
  return 0.5f * x * (1.0f + erff(x * 0.70710678118654752f));
}

// CDNA5 async copy: global -> LDS, 16B per lane, tracked by ASYNCcnt
__device__ __forceinline__ void async_b128(unsigned lds_off, const void* gaddr) {
  asm volatile("global_load_async_to_lds_b128 %0, %1, off"
               :: "v"(lds_off), "v"(gaddr) : "memory");
}
__device__ __forceinline__ void wait_async0() {
  asm volatile("s_wait_asynccnt 0" ::: "memory");
}
__device__ __forceinline__ unsigned lds_off_of(const void* p) {
  return (unsigned)(unsigned long long)p;   // low 32 bits of flat addr = LDS offset
}

// ---------------------------------------------------------------------------
// bf16 WMMA GEMM: C[m][n] = sum_k A[m][k] * BT[n][k]   (both operands bf16)
// block tile M=64 x N=128, K-step 32; 8 waves, each wave 2x2 16x16 subtiles.
// Double-buffered LDS staging via global_load_async_to_lds_b128: the async
// copy for step i+1 overlaps the 4 WMMAs of step i. One barrier per step.
// grid = (N/128, M/64, batch)
// ---------------------------------------------------------------------------
template<bool COL, bool ROW, bool GELU, bool RES, bool F32OUT, bool BFOUT>
__global__ __launch_bounds__(256) void gemm_bf16_kernel(
    const __bf16* __restrict__ A,  long lda,  long sAb,
    const __bf16* __restrict__ BT, long ldbt, long sBb,
    float*        __restrict__ Cm, long ldc,  long sCb,
    __bf16*       __restrict__ Cbf,
    int K,
    const float* __restrict__ colBias,
    const float* __restrict__ rowBias,
    const float* __restrict__ resid, long ldres)
{
  __shared__ __bf16 As[2][64][40];    // rows 80B -> every 16B chunk aligned
  __shared__ __bf16 Bs[2][128][40];   // [n][k]

  const int t  = threadIdx.x;
  const int bz = blockIdx.z;
  const int n0 = blockIdx.x * 128;
  const int m0 = blockIdx.y * 64;
  A  += (long)bz * sAb;
  BT += (long)bz * sBb;

  const int wave = t >> 5, lane = t & 31;
  const int wmh  = (wave & 1) * 32;     // m half
  const int wnq  = (wave >> 1) * 32;    // n quarter
  const int lrow = lane & 15;
  const int loff = (lane >> 4) * 8;

  v8f a00 = {}, a01 = {}, a10 = {}, a11 = {};

  // staging assignments (16B chunks)
  const int ar = t >> 2, ak = (t & 3) * 8;        // A: 64 rows x 32k
  const int br = t >> 1, bk = (t & 1) * 16;       // B: 128 rows x 32k
  const __bf16* apg = A  + (long)(m0 + ar) * lda  + ak;
  const __bf16* bpg = BT + (long)(n0 + br) * ldbt + bk;
  const unsigned lA[2]  = { lds_off_of(&As[0][ar][ak]),     lds_off_of(&As[1][ar][ak]) };
  const unsigned lB0[2] = { lds_off_of(&Bs[0][br][bk]),     lds_off_of(&Bs[1][br][bk]) };
  const unsigned lB1[2] = { lds_off_of(&Bs[0][br][bk + 8]), lds_off_of(&Bs[1][br][bk + 8]) };

  // prologue: stage k0 = 0 into buffer 0
  async_b128(lA[0],  apg);
  async_b128(lB0[0], bpg);
  async_b128(lB1[0], bpg + 8);

  int cur = 0;
  for (int k0 = 0; k0 < K; k0 += 32, cur ^= 1) {
    wait_async0();          // my writes to buf[cur] done
    __syncthreads();        // everyone's writes done; everyone's reads of buf[cur^1] done
    if (k0 + 32 < K) {      // overlap next stage with this step's WMMAs
      const int nxt = cur ^ 1;
      async_b128(lA[nxt],  apg + k0 + 32);
      async_b128(lB0[nxt], bpg + k0 + 32);
      async_b128(lB1[nxt], bpg + k0 + 40);
    }

    union { v16bf v; unsigned u[8]; } fa0, fa1, fb0, fb1;
    #pragma unroll
    for (int j = 0; j < 8; ++j) {
      const int kb = ((j < 4) ? (2 * j) : (8 + 2 * j)) + loff;
      fa0.u[j] = *(const unsigned*)&As[cur][wmh + lrow][kb];
      fa1.u[j] = *(const unsigned*)&As[cur][wmh + 16 + lrow][kb];
      fb0.u[j] = *(const unsigned*)&Bs[cur][wnq + lrow][kb];
      fb1.u[j] = *(const unsigned*)&Bs[cur][wnq + 16 + lrow][kb];
    }
    a00 = __builtin_amdgcn_wmma_f32_16x16x32_bf16(false, fa0.v, false, fb0.v, (short)0, a00, false, false);
    a01 = __builtin_amdgcn_wmma_f32_16x16x32_bf16(false, fa0.v, false, fb1.v, (short)0, a01, false, false);
    a10 = __builtin_amdgcn_wmma_f32_16x16x32_bf16(false, fa1.v, false, fb0.v, (short)0, a10, false, false);
    a11 = __builtin_amdgcn_wmma_f32_16x16x32_bf16(false, fa1.v, false, fb1.v, (short)0, a11, false, false);
  }

  if (F32OUT) Cm  += (long)bz * sCb;
  if (BFOUT)  Cbf += (long)bz * sCb;
  const int hi8 = (lane >> 4) * 8;
  #pragma unroll
  for (int mi = 0; mi < 2; ++mi) {
    #pragma unroll
    for (int ni = 0; ni < 2; ++ni) {
      const v8f acc = (mi == 0) ? (ni == 0 ? a00 : a01) : (ni == 0 ? a10 : a11);
      const int n = n0 + wnq + ni * 16 + (lane & 15);
      const float cb = COL ? colBias[n] : 0.0f;
      #pragma unroll
      for (int r = 0; r < 8; ++r) {
        const int m = m0 + wmh + mi * 16 + hi8 + r;
        float v = acc[r] + cb;
        if (ROW)  v += rowBias[m];
        if (GELU) v = geluf(v);
        if (RES)  v += resid[(long)m * ldres + n];
        if (F32OUT) Cm[(long)m * ldc + n]  = v;
        if (BFOUT)  Cbf[(long)m * ldc + n] = f2bf(v);
      }
    }
  }
}

// one-shot convert f32 -> bf16 (same layout)
__global__ void cvt_kernel(const float* __restrict__ in, __bf16* __restrict__ out) {
  const long gid = (long)blockIdx.x * 256 + threadIdx.x;
  out[gid] = f2bf(in[gid]);
}

// one-shot tiled transpose+convert: out[c][r] = bf16(in[r][c]); grid (R/64, C/64)
__global__ __launch_bounds__(256) void tconv_kernel(
    const float* __restrict__ in, long ldin,
    __bf16* __restrict__ out, long ldout)
{
  __shared__ __bf16 tile[64][65];
  const int r0 = blockIdx.x * 64, c0 = blockIdx.y * 64;
  const int t = threadIdx.x;
  const int lr = t >> 2, lc = (t & 3) * 16;
  #pragma unroll
  for (int i = 0; i < 16; ++i)
    tile[lr][lc + i] = f2bf(in[(long)(r0 + lr) * ldin + c0 + lc + i]);
  __syncthreads();
  #pragma unroll
  for (int i = 0; i < 16; ++i)
    out[(long)(c0 + lr) * ldout + r0 + lc + i] = tile[lc + i][lr];
}

// big layernorm stats per batch over len contiguous elements
__global__ __launch_bounds__(256) void bigln_stats_kernel(
    const float* __restrict__ x, long len, float* __restrict__ stats)
{
  __shared__ float ssum[256], ssq[256];
  const int b = blockIdx.x, t = threadIdx.x;
  const float* p = x + (long)b * len;
  float s = 0.f, q2 = 0.f;
  for (long i = t; i < len; i += 256) { const float v = p[i]; s += v; q2 += v * v; }
  ssum[t] = s; ssq[t] = q2; __syncthreads();
  for (int st = 128; st > 0; st >>= 1) {
    if (t < st) { ssum[t] += ssum[t + st]; ssq[t] += ssq[t + st]; }
    __syncthreads();
  }
  if (t == 0) {
    const float m   = ssum[0] / (float)len;
    const float var = ssq[0] / (float)len - m * m;
    stats[2 * b] = m; stats[2 * b + 1] = rsqrtf(var + 1e-5f);
  }
}

__global__ void apply_ln_kernel(const float* __restrict__ x, float* __restrict__ o,
    const float* __restrict__ stats, long len,
    const float* __restrict__ g, const float* __restrict__ bt)
{
  const long gid = (long)blockIdx.x * 256 + threadIdx.x;
  const long b = gid / len, r = gid % len;
  const float m = stats[2 * b], rs = stats[2 * b + 1];
  o[gid] = (x[gid] - m) * rs * g[r] + bt[r];
}

// depthwise 3x3 conv (pad 1) + window repack; writes f32 wbuf and bf16 wbufT
__global__ void conv_win_kernel(const float* __restrict__ h1,
    const float* __restrict__ wc, const float* __restrict__ bc,
    float* __restrict__ wbuf, __bf16* __restrict__ wbufT)
{
  const int gid = blockIdx.x * 256 + threadIdx.x;
  const int w_ = gid & 63, h = (gid >> 6) & 63, c = (gid >> 12) & 127, b = gid >> 19;
  const float* base = h1 + (((long)b * 128 + c) << 12);
  float s = bc[c];
  #pragma unroll
  for (int dy = -1; dy <= 1; ++dy)
    #pragma unroll
    for (int dx = -1; dx <= 1; ++dx) {
      const int hh = h + dy, ww = w_ + dx;
      if (hh >= 0 && hh < 64 && ww >= 0 && ww < 64)
        s += wc[c * 9 + (dy + 1) * 3 + (dx + 1)] * base[hh * 64 + ww];
    }
  const int nwin = (((h >> 3) * 8 + (w_ >> 3)) << 6) + ((h & 7) * 8 + (w_ & 7));
  wbuf[(((long)b * 128 + c) << 12) + nwin] = s;
  wbufT[(((long)b << 12) + nwin) * 128 + c] = f2bf(s);
}

// skinny projection (q/k, O=16)
__global__ void skinny_proj_kernel(const float* __restrict__ Wp,
    const float* __restrict__ bias, const float* __restrict__ wbuf,
    float* __restrict__ outp, int O)
{
  const int gid = blockIdx.x * 256 + threadIdx.x;
  const int n = gid & 4095;
  const int o = (gid >> 12) % O;
  const int b = gid / (O * 4096);
  const float* col = wbuf + (((long)b * 128) << 12) + n;
  float s = bias[o];
  #pragma unroll 4
  for (int c = 0; c < 128; ++c) s += Wp[o * 128 + c] * col[(long)c << 12];
  outp[((long)(b * O + o) << 12) + n] = s;
}

// attention pass 1: per-row max and 1/sum(exp) (wave per row)
__global__ __launch_bounds__(256) void attn_stats_kernel(
    const float* __restrict__ q, const float* __restrict__ k,
    float* __restrict__ mx, float* __restrict__ si)
{
  const int t = threadIdx.x;
  const int row = blockIdx.x * 8 + (t >> 5);
  const int lane = t & 31;
  const int b = row >> 12, n = row & 4095;
  const long qb = (long)b * 16 * 4096;
  float qv[16];
  #pragma unroll
  for (int c = 0; c < 16; ++c) qv[c] = q[qb + ((long)c << 12) + n];
  float lmax = -3.4e38f;
  for (int m = lane; m < 4096; m += 32) {
    float e = 0.f;
    #pragma unroll
    for (int c = 0; c < 16; ++c) e += qv[c] * k[qb + ((long)c << 12) + m];
    lmax = fmaxf(lmax, e);
  }
  #pragma unroll
  for (int s = 16; s > 0; s >>= 1) lmax = fmaxf(lmax, __shfl_xor(lmax, s, 32));
  float lsum = 0.f;
  for (int m = lane; m < 4096; m += 32) {
    float e = 0.f;
    #pragma unroll
    for (int c = 0; c < 16; ++c) e += qv[c] * k[qb + ((long)c << 12) + m];
    lsum += __expf(e - lmax);
  }
  #pragma unroll
  for (int s = 16; s > 0; s >>= 1) lsum += __shfl_xor(lsum, s, 32);
  if (lane == 0) { mx[row] = lmax; si[row] = 1.0f / lsum; }
}

// attention pass 2: block = (b, 32-token n-tile); wave w owns channels [16w,16w+16)
// out_w[b][c][n] = gamma * sum_m v[c][m]*p[n][m] + wbuf[b][c][n]
// v tile double-buffered via async-to-LDS; overlaps with p-tile VALU work.
__global__ __launch_bounds__(256) void attn_out_kernel(
    const float* __restrict__ q, const float* __restrict__ k,
    const __bf16* __restrict__ vbf, const float* __restrict__ wbuf,
    const float* __restrict__ mx, const float* __restrict__ si,
    const float* __restrict__ gamma_p, float* __restrict__ out_w)
{
  __shared__ float  q_s[16][33];
  __shared__ float  k_s[16][33];
  __shared__ __bf16 v_s[2][128][40];
  __shared__ __bf16 p_s[32][40];
  __shared__ float  mx_s[32], si_s[32];

  const int t  = threadIdx.x;
  const int n0 = blockIdx.x * 32;
  const int b  = blockIdx.y;
  const long qbase = (long)b * 16 * 4096;
  const long vbase = (long)b * 128 * 4096;

  #pragma unroll
  for (int i = 0; i < 2; ++i) {
    const int idx = i * 256 + t, c = idx >> 5, n = idx & 31;
    q_s[c][n] = q[qbase + ((long)c << 12) + n0 + n];
  }
  if (t < 32) { mx_s[t] = mx[(long)b * 4096 + n0 + t];
                si_s[t] = si[(long)b * 4096 + n0 + t]; }

  const int wave = t >> 5, lane = t & 31;
  const int lrow = lane & 15, loff = (lane >> 4) * 8;
  v8f acc0 = {}, acc1 = {};

  // v staging: 2 chunks of 16B per thread per step
  const int vc = t >> 1, vp = (t & 1) * 2;
  const unsigned lV0[2] = { lds_off_of(&v_s[0][vc][vp * 8]),
                            lds_off_of(&v_s[1][vc][vp * 8]) };
  const unsigned lV1[2] = { lds_off_of(&v_s[0][vc][(vp + 1) * 8]),
                            lds_off_of(&v_s[1][vc][(vp + 1) * 8]) };
  const __bf16* vpg = vbf + vbase + ((long)vc << 12) + vp * 8;
  const float*  kpg = k + qbase + (((long)(t >> 5)) << 12) + (t & 31);

  async_b128(lV0[0], vpg);
  async_b128(lV1[0], vpg + 8);

  int cur = 0;
  for (int m0 = 0; m0 < 4096; m0 += 32, cur ^= 1) {
    // current k tile into registers (overlaps async wait)
    float kr0 = kpg[m0];                 // c = t>>5,      m = t&31
    float kr1 = kpg[m0 + (8L << 12)];    // c = 8 + t>>5
    wait_async0();
    __syncthreads();
    if (m0 + 32 < 4096) {
      const int nxt = cur ^ 1;
      async_b128(lV0[nxt], vpg + m0 + 32);
      async_b128(lV1[nxt], vpg + m0 + 40);
    }
    k_s[t >> 5][t & 31] = kr0;
    k_s[8 + (t >> 5)][t & 31] = kr1;
    __syncthreads();
    #pragma unroll
    for (int i = 0; i < 4; ++i) {                       // p tile 32n x 32m
      const int idx = i * 256 + t, n = idx >> 5, mm = idx & 31;
      float e = 0.f;
      #pragma unroll
      for (int c = 0; c < 16; ++c) e += q_s[c][n] * k_s[c][mm];
      p_s[n][mm] = f2bf(__expf(e - mx_s[n]) * si_s[n]);
    }
    __syncthreads();
    union { v16bf v; unsigned u[8]; } fa, fb0, fb1;
    #pragma unroll
    for (int j = 0; j < 8; ++j) {
      const int kb = ((j < 4) ? (2 * j) : (8 + 2 * j)) + loff;
      fa.u[j]  = *(const unsigned*)&v_s[cur][wave * 16 + lrow][kb];
      fb0.u[j] = *(const unsigned*)&p_s[lrow][kb];
      fb1.u[j] = *(const unsigned*)&p_s[16 + lrow][kb];
    }
    acc0 = __builtin_amdgcn_wmma_f32_16x16x32_bf16(false, fa.v, false, fb0.v, (short)0, acc0, false, false);
    acc1 = __builtin_amdgcn_wmma_f32_16x16x32_bf16(false, fa.v, false, fb1.v, (short)0, acc1, false, false);
  }

  const float g = gamma_p[0];
  const int chi = (lane >> 4) * 8;
  #pragma unroll
  for (int ni = 0; ni < 2; ++ni) {
    const v8f acc = ni ? acc1 : acc0;
    const int n = n0 + ni * 16 + (lane & 15);
    #pragma unroll
    for (int r = 0; r < 8; ++r) {
      const int c = wave * 16 + r + chi;
      const long idx = vbase + ((long)c << 12) + n;
      out_w[idx] = g * acc[r] + wbuf[idx];
    }
  }
}

// scramble-reorder + residual: t[b][h*64+w][c] = out_w[b][c][w*64+h] + x[b][c][h][w]
__global__ void build_t_kernel(const float* __restrict__ out_w,
    const float* __restrict__ x, float* __restrict__ tb)
{
  const int gid = blockIdx.x * 256 + threadIdx.x;     // (b,h,w,c)
  const int c = gid & 127, w_ = (gid >> 7) & 63, h = (gid >> 13) & 63, b = gid >> 19;
  const long src = (((long)b * 128 + c) << 12);
  tb[gid] = out_w[src + (w_ * 64 + h)] + x[src + h * 64 + w_];
}

// per-row layernorm (wave per row); outputs: bf16 row-major and/or bf16 transposed
__global__ __launch_bounds__(256) void rowln_kernel(
    const float* __restrict__ in, int ld, int D,
    const float* __restrict__ g, const float* __restrict__ bt,
    __bf16* __restrict__ outRM,
    __bf16* __restrict__ outT, long ldT, int rowsPerB, long strideT)
{
  const int row = blockIdx.x * 8 + (threadIdx.x >> 5);
  const int lane = threadIdx.x & 31;
  const float* p = in + (long)row * ld;
  float s = 0.f, q2 = 0.f;
  for (int i = lane; i < D; i += 32) { const float v = p[i]; s += v; q2 += v * v; }
  #pragma unroll
  for (int st = 16; st > 0; st >>= 1) {
    s += __shfl_xor(s, st, 32); q2 += __shfl_xor(q2, st, 32);
  }
  const float m  = s / (float)D;
  const float rs = rsqrtf(q2 / (float)D - m * m + 1e-5f);
  for (int i = lane; i < D; i += 32) {
    const float v = (p[i] - m) * rs * g[i] + bt[i];
    if (outRM) outRM[(long)row * D + i] = f2bf(v);
    if (outT) {
      const long bb = row / rowsPerB, mm = row % rowsPerB;
      outT[bb * strideT + (long)i * ldT + mm] = f2bf(v);
    }
  }
}

// elementwise u .* sp -> bf16 (A operand of final GEMM)
__global__ void prod_kernel(const float* __restrict__ z, const float* __restrict__ sp,
                            __bf16* __restrict__ pbf)
{
  const long gid = (long)blockIdx.x * 256 + threadIdx.x;   // 16384 x 512
  const long m = gid >> 9, k = gid & 511;
  pbf[gid] = f2bf(z[m * 1024 + k] * sp[gid]);
}

// ---------------------------------------------------------------------------
extern "C" void kernel_launch(void* const* d_in, const int* in_sizes, int n_in,
                              void* d_out, int out_size, void* d_ws, size_t ws_size,
                              hipStream_t stream) {
  const float* x     = (const float*)d_in[0];
  const float* ln1_g = (const float*)d_in[1];
  const float* ln1_b = (const float*)d_in[2];
  const float* dw_w  = (const float*)d_in[3];
  const float* dw_b  = (const float*)d_in[4];
  const float* qw    = (const float*)d_in[5];
  const float* qbias = (const float*)d_in[6];
  const float* kw    = (const float*)d_in[7];
  const float* kbias = (const float*)d_in[8];
  const float* vw    = (const float*)d_in[9];
  const float* vbias = (const float*)d_in[10];
  const float* att_g = (const float*)d_in[11];
  const float* ln2_g = (const float*)d_in[12];
  const float* ln2_b = (const float*)d_in[13];
  const float* gln_g = (const float*)d_in[14];
  const float* gln_b = (const float*)d_in[15];
  const float* W1    = (const float*)d_in[16];
  const float* b1    = (const float*)d_in[17];
  const float* sgu_g = (const float*)d_in[18];
  const float* sgu_b = (const float*)d_in[19];
  const float* Wsp   = (const float*)d_in[20];
  const float* bsp   = (const float*)d_in[21];
  const float* W2    = (const float*)d_in[22];
  const float* b2    = (const float*)d_in[23];
  float* out = (float*)d_out;

  float* ws = (float*)d_ws;
  // f32 regions
  float* h1   = ws;                         // 2,097,152 (reused for out_w)
  float* wbuf = ws + 2097152L;
  float* qb_  = ws + 4194304L;              // 262,144
  float* kb_  = ws + 4456448L;
  float* tb   = ws + 4718592L;              // 2,097,152 (t_pre -> r2 in place)
  float* z    = ws + 6815744L;              // 16,777,216
  float* sp   = ws + 23592960L;             // 8,388,608
  float* mxs  = ws + 31981568L;
  float* sis  = ws + 31997952L;
  float* st1  = ws + 32014336L;
  float* st2  = ws + 32014344L;
  float* oww  = h1;                         // out_w reuses h1 (free after conv)
  // bf16 regions (16B aligned)
  __bf16* bb      = (__bf16*)(ws + 32014352L);
  __bf16* wbufT   = bb;                     // 2,097,152 halves  [b][n][c]
  __bf16* vbbf    = bb + 2097152L;          // 2,097,152         [b][c][n]
  __bf16* tgbf    = bb + 4194304L;          // 2,097,152         [row][128]
  __bf16* W1T     = bb + 6291456L;          // 131,072           [n][k]
  __bf16* Wspbf   = bb + 6422528L;          // 16,777,216        [n][m]
  __bf16* vglnT   = bb + 23199744L;         // 8,388,608         [b][d][m]
  __bf16* W2T     = bb + 31588352L;         // 65,536            [n][k]
  __bf16* vwbf    = bb + 31653888L;         // 16,384            [o][c]
  __bf16* prodbf  = bb + 31670272L;         // 8,388,608         [row][512]

  // one-time weight conversions (L2-resident, trivial bandwidth)
  cvt_kernel<<<64, 256, 0, stream>>>(vw, vwbf);
  tconv_kernel<<<dim3(2, 16), 256, 0, stream>>>(W1, 1024, W1T, 128);
  cvt_kernel<<<65536, 256, 0, stream>>>(Wsp, Wspbf);
  tconv_kernel<<<dim3(8, 2), 256, 0, stream>>>(W2, 128, W2T, 512);

  // 1) LN1 (per batch over C*H*W)
  bigln_stats_kernel<<<4, 256, 0, stream>>>(x, 524288L, st1);
  apply_ln_kernel<<<8192, 256, 0, stream>>>(x, h1, st1, 524288L, ln1_g, ln1_b);

  // 2) depthwise conv + window repack (f32 + bf16-transposed copies)
  conv_win_kernel<<<8192, 256, 0, stream>>>(h1, dw_w, dw_b, wbuf, wbufT);

  // 3) q/k skinny projections; v projection via WMMA (bf16 output, row bias)
  skinny_proj_kernel<<<1024, 256, 0, stream>>>(qw, qbias, wbuf, qb_, 16);
  skinny_proj_kernel<<<1024, 256, 0, stream>>>(kw, kbias, wbuf, kb_, 16);
  gemm_bf16_kernel<false, true, false, false, false, true>
      <<<dim3(32, 2, 4), 256, 0, stream>>>(
      vwbf, 128, 0, wbufT, 128, 524288L, nullptr, 4096, 524288L, vbbf,
      128, nullptr, vbias, nullptr, 0);

  // 4) attention
  attn_stats_kernel<<<2048, 256, 0, stream>>>(qb_, kb_, mxs, sis);
  attn_out_kernel<<<dim3(128, 4), 256, 0, stream>>>(qb_, kb_, vbbf, wbuf,
                                                    mxs, sis, att_g, oww);

  // 5) scramble-reorder + residual -> (B,N,C); LN2 (in place); gLN -> bf16 tg
  build_t_kernel<<<8192, 256, 0, stream>>>(oww, x, tb);
  bigln_stats_kernel<<<4, 256, 0, stream>>>(tb, 524288L, st2);
  apply_ln_kernel<<<8192, 256, 0, stream>>>(tb, tb, st2, 524288L, ln2_g, ln2_b);
  rowln_kernel<<<2048, 256, 0, stream>>>(tb, 128, 128, gln_g, gln_b,
                                         tgbf, nullptr, 0, 0, 0);

  // 6) z = gelu(tg @ W1 + b1)   (16384 x 1024, K=128); col bias + gelu
  gemm_bf16_kernel<true, false, true, false, true, false>
      <<<dim3(8, 256, 1), 256, 0, stream>>>(
      tgbf, 128, 0, W1T, 128, 0, z, 1024, 0, nullptr,
      128, b1, nullptr, nullptr, 0);

  // 7) row-LN of vg half (reads z[:,512:], writes bf16 transposed per batch)
  rowln_kernel<<<2048, 256, 0, stream>>>(z + 512, 1024, 512, sgu_g, sgu_b,
                                         nullptr, vglnT, 4096, 4096, 2097152L);

  // 8) spatial gating: sp[b] = Wsp @ vg[b] + bsp[row]  (4096x512, K=4096, x4)
  gemm_bf16_kernel<false, true, false, false, true, false>
      <<<dim3(4, 64, 4), 256, 0, stream>>>(
      Wspbf, 4096, 0, vglnT, 4096, 2097152L, sp, 512, 2097152L, nullptr,
      4096, nullptr, bsp, nullptr, 0);

  // 9) prod = bf16(u .* sp); out = prod @ W2 + b2 + r2 -> (B,H,W,C)
  prod_kernel<<<32768, 256, 0, stream>>>(z, sp, prodbf);
  gemm_bf16_kernel<true, false, false, true, true, false>
      <<<dim3(1, 256, 1), 256, 0, stream>>>(
      prodbf, 512, 0, W2T, 512, 0, out, 128, 0, nullptr,
      512, b2, nullptr, tb, 128);
}
